// DeformConv_30940944400805
// MI455X (gfx1250) — matrix operand order repaired and checked by reading, not compiled, not run on hardware
//
#include <hip/hip_runtime.h>

typedef _Float16 v16h __attribute__((ext_vector_type(16)));
typedef _Float16 v8h  __attribute__((ext_vector_type(8)));
typedef float    v8f  __attribute__((ext_vector_type(8)));

constexpr int TN      = 128;            // output columns per block
constexpr int CIN     = 32;
constexpr int KTAPS   = 27;             // 3*3*3
constexpr int DSZ     = 48;
constexpr int NTOT    = DSZ * DSZ * DSZ;   // 110592
constexpr int BSTRIDE = 24;             // halves per B-fragment lane row (16 + 8 pad)

// Fused deformable-im2col + WMMA GEMM.
// out[32, N] = W[32, 32*27] * cols[32*27, N] + bias, cols built per-tap in LDS.
__global__ __launch_bounds__(256)
void deform_conv3d_wmma(const float* __restrict__ x,
                        const float* __restrict__ off,
                        const float* __restrict__ wgt,
                        const float* __restrict__ bias,
                        float* __restrict__ out)
{
    // A fragments for ALL taps, packed once: [k=27][mh=2][lane=32] of v16h (54 KB)
    __shared__ v16h Aall[KTAPS * 2 * 32];
    // B fragments: [nt=8][lane=32][BSTRIDE halves] (32x16 f16 B-tiles, padded)
    __shared__ __align__(16) _Float16 Bfrag[8 * 32 * BSTRIDE];
    // trilinear corner weights / clamped flat indices, shared across channels
    __shared__ float cw[8][TN];
    __shared__ int   cidx[8][TN];

    const int t        = threadIdx.x;
    const int wave     = t >> 5;        // 0..7 -> owns 16-column tile `wave`
    const int lane     = t & 31;
    const int tileBase = blockIdx.x * TN;

    // ---- one-time: pack weight (f32) -> f16 WMMA A-fragment order, all taps ----
    // A 16x32 f16 layout: lane = laneHalf*16 + m; K = (j>>3)*16 + laneHalf*8 + (j&7)
    {
        _Float16* Af = (_Float16*)Aall;
        for (int e = t; e < KTAPS * 1024; e += 256) {
            const int k  = e >> 10;
            const int j  = e & 15;
            const int ln = (e >> 4) & 31;
            const int mh = (e >> 9) & 1;
            const int co = mh * 16 + (ln & 15);
            const int ci = ((j >> 3) << 4) | (((ln >> 4) & 1) << 3) | (j & 7);
            Af[e] = (_Float16)wgt[(co * CIN + ci) * KTAPS + k];
        }
    }
    __syncthreads();

    // ---- loop-invariant per-column geometry (hoisted out of the tap loop) ----
    const int  myn   = t & (TN - 1);       // column this thread gathers for
    const int  nglob = tileBase + t;       // column this thread does interp for
    const int  wo    = nglob % DSZ;
    const int  ho    = (nglob / DSZ) % DSZ;
    const int  dn    = nglob / (DSZ * DSZ);
    const float baseD = (float)(dn - 1);   // pad=1, stride=1, dil=1
    const float baseH = (float)(ho - 1);
    const float baseW = (float)(wo - 1);
    const float* offD = off + 0 * KTAPS * NTOT + nglob;
    const float* offH = off + 1 * KTAPS * NTOT + nglob;
    const float* offW = off + 2 * KTAPS * NTOT + nglob;
    const int  chBase = (t >> 7) << 4;     // B lane-half bit (0/16); channel offset is chBase>>1
    const int  myNt   = myn >> 4;
    const int  myNl   = myn & 15;
    // this thread's full 32-byte B-fragment lane row (j = half*8 + jj spans 0..15)
    v8h* browp = (v8h*)&Bfrag[(myNt * 32 + ((chBase & 16) | myNl)) * BSTRIDE];

    v8f acc0 = {0.f, 0.f, 0.f, 0.f, 0.f, 0.f, 0.f, 0.f};  // Co 0..15
    v8f acc1 = acc0;                                        // Co 16..31

    for (int k = 0; k < KTAPS; ++k) {
        const int kd = k / 9, kh = (k / 3) % 3, kw = k % 3;

        // ---- interpolation params for the TN columns of this tile ----
        // (Overlaps the previous tap's WMMA: writes cw/cidx only.)
        if (t < TN) {
            // offset layout: [3][K][N]; streamed exactly once -> non-temporal
            const float od = __builtin_nontemporal_load(offD + k * NTOT);
            const float oh = __builtin_nontemporal_load(offH + k * NTOT);
            const float ow = __builtin_nontemporal_load(offW + k * NTOT);
            const float pd = baseD + (float)kd + od;
            const float ph = baseH + (float)kh + oh;
            const float pw = baseW + (float)kw + ow;
            const float fd0 = floorf(pd), fh0 = floorf(ph), fw0 = floorf(pw);
            const int   d0 = (int)fd0, h0 = (int)fh0, w0 = (int)fw0;
            const float fd = pd - fd0, fh = ph - fh0, fw = pw - fw0;
            const float wdv[2] = {1.f - fd, fd};
            const float whv[2] = {1.f - fh, fh};
            const float wwv[2] = {1.f - fw, fw};
            #pragma unroll
            for (int a = 0; a < 2; ++a) {
                const int  di  = d0 + a;
                const bool vd  = (di >= 0) && (di < DSZ);
                const int  dic = min(max(di, 0), DSZ - 1);
                #pragma unroll
                for (int b2 = 0; b2 < 2; ++b2) {
                    const int  hi  = h0 + b2;
                    const bool vh  = (hi >= 0) && (hi < DSZ);
                    const int  hic = min(max(hi, 0), DSZ - 1);
                    #pragma unroll
                    for (int c2 = 0; c2 < 2; ++c2) {
                        const int  wi  = w0 + c2;
                        const bool vw  = (wi >= 0) && (wi < DSZ);
                        const int  wic = min(max(wi, 0), DSZ - 1);
                        const int  q   = a * 4 + b2 * 2 + c2;
                        cw[q][t]   = (vd && vh && vw) ? wdv[a] * whv[b2] * wwv[c2] : 0.f;
                        cidx[q][t] = (dic * DSZ + hic) * DSZ + wic;
                    }
                }
            }
        }
        __syncthreads();   // cw/cidx ready; prev-tap Bfrag consumers already done

        // ---- gather + trilinear blend, pack B fragments (f16) ----
        // Thread owns column myn and channels (chBase>>1)+0..7 and 16+(chBase>>1)+0..7,
        // i.e. one contiguous 32B B-fragment lane row -> two ds_store_b128.
        {
            float w8[8]; int i8[8];
            #pragma unroll
            for (int q = 0; q < 8; ++q) { w8[q] = cw[q][myn]; i8[q] = cidx[q][myn]; }
            #pragma unroll
            for (int half = 0; half < 2; ++half) {
                v8h pack;
                #pragma unroll
                for (int jj = 0; jj < 8; ++jj) {
                    const int c = half * 16 + (chBase >> 1) + jj;
                    const float* xc = x + c * NTOT;
                    float v = 0.f;
                    #pragma unroll
                    for (int q = 0; q < 8; ++q) v += w8[q] * xc[i8[q]];
                    pack[jj] = (_Float16)v;
                }
                browp[half] = pack;    // 16B aligned (row stride 48B)
            }
        }
        __syncthreads();   // Bfrag ready

        // ---- WMMA: wave handles column-tile `wave`, both Co halves ----
        // Fragment ds_loads are dscnt-waited before the WMMAs, hence complete
        // before this wave reaches the next barrier -> safe vs tap k+1 writers.
        {
            const v16h a0 = Aall[k * 64 + lane];
            const v16h a1 = Aall[k * 64 + 32 + lane];
            const v8h* bp = (const v8h*)&Bfrag[(wave * 32 + lane) * BSTRIDE];
            const v8h blo = bp[0];
            const v8h bhi = bp[1];
            v16h b;
            #pragma unroll
            for (int j = 0; j < 8; ++j) { b[j] = blo[j]; b[8 + j] = bhi[j]; }
            acc0 = __builtin_amdgcn_wmma_f32_16x16x32_f16(
                false, a0, false, b, (short)0, acc0, false, false);
            acc1 = __builtin_amdgcn_wmma_f32_16x16x32_f16(
                false, a1, false, b, (short)0, acc1, false, false);
        }
    }

    // ---- epilogue: bias + store (C/D layout: VGPR r -> M = r + 8*(lane>>4)) ----
    const int nl   = lane & 15;
    const int lh   = lane >> 4;
    const int nout = tileBase + wave * 16 + nl;
    #pragma unroll
    for (int r = 0; r < 8; ++r) {
        const int m = r + lh * 8;
        out[m * NTOT + nout]        = acc0[r] + bias[m];
        out[(16 + m) * NTOT + nout] = acc1[r] + bias[16 + m];
    }
}

extern "C" void kernel_launch(void* const* d_in, const int* in_sizes, int n_in,
                              void* d_out, int out_size, void* d_ws, size_t ws_size,
                              hipStream_t stream) {
    const float* x    = (const float*)d_in[0];   // [1,32,48,48,48]
    const float* off  = (const float*)d_in[1];   // [1,81,48,48,48]
    const float* wgt  = (const float*)d_in[2];   // [32,32,3,3,3]
    const float* bias = (const float*)d_in[3];   // [32]
    float* out = (float*)d_out;                  // [1,32,48,48,48]

    dim3 grid(NTOT / TN);   // 864 tiles, N divisible by TN exactly
    dim3 block(256);
    hipLaunchKernelGGL(deform_conv3d_wmma, grid, block, 0, stream,
                       x, off, wgt, bias, out);
}
